// MoE_7378753814908
// MI455X (gfx1250) — matrix operand order repaired and checked by reading, not compiled, not run on hardware
//
#include <hip/hip_runtime.h>
#include <cstdint>

#define E_   16
#define D_   2048
#define I_   1408
#define CAP_ 1024
#define LDA  40   // padded LDS row stride (bf16 elems): 80 bytes, keeps 16B alignment

typedef __attribute__((ext_vector_type(8)))  float  v8f;
typedef __attribute__((ext_vector_type(8)))  __bf16 v8bf;
typedef __attribute__((ext_vector_type(16))) __bf16 v16bf;
typedef __attribute__((ext_vector_type(4)))  int    v4i;

// ---------- bf16 conversion helpers ----------
static __device__ __forceinline__ unsigned int f2bf(float f) {
  unsigned int u = __builtin_bit_cast(unsigned int, f);
  u += 0x7FFFu + ((u >> 16) & 1u);        // round-to-nearest-even
  return u >> 16;                          // bf16 bits in low 16
}

#if __has_builtin(__builtin_amdgcn_cvt_pk_bf16_f32)
typedef __attribute__((ext_vector_type(2))) __bf16 v2bf;
static __device__ __forceinline__ unsigned int pack2(float a, float b) {
  v2bf t = __builtin_amdgcn_cvt_pk_bf16_f32(a, b);   // single v_cvt_pk_bf16_f32
  return __builtin_bit_cast(unsigned int, t);
}
#else
static __device__ __forceinline__ unsigned int pack2(float a, float b) {
  return f2bf(a) | (f2bf(b) << 16);
}
#endif

static __device__ __forceinline__ uint4 cvt8(const float* p) {
  float4 f0 = *(const float4*)p;
  float4 f1 = *(const float4*)(p + 4);
  uint4 r;
  r.x = pack2(f0.x, f0.y);
  r.y = pack2(f0.z, f0.w);
  r.z = pack2(f1.x, f1.y);
  r.w = pack2(f1.z, f1.w);
  return r;
}

// ---------- async global->LDS copy (CDNA5 ASYNCcnt path), guarded ----------
#if __has_builtin(__builtin_amdgcn_global_load_async_to_lds_b128)
#define ASYNC_LDS 1
#define AS_GLOBAL __attribute__((address_space(1)))
#define AS_LOCAL  __attribute__((address_space(3)))
static __device__ __forceinline__ void async_copy16(const unsigned short* g,
                                                    unsigned short* l) {
  __builtin_amdgcn_global_load_async_to_lds_b128((AS_GLOBAL v4i*)g,
                                                 (AS_LOCAL v4i*)l, 0, 0);
}
static __device__ __forceinline__ void async_wait0() {
#if __has_builtin(__builtin_amdgcn_s_wait_asynccnt)
  __builtin_amdgcn_s_wait_asynccnt(0);
#else
  asm volatile("s_wait_asynccnt 0" ::: "memory");
#endif
}
#endif

// ---------- WMMA fragment helpers ----------
static __device__ __forceinline__ v16bf mk16(v8bf lo, v8bf hi) {
  v16bf a;
#pragma unroll
  for (int i = 0; i < 8; ++i) { a[i] = lo[i]; a[i + 8] = hi[i]; }
  return a;
}
// A fragment: 16x32 bf16, lanes 0-15 hold M, K chunks {8*half, 16+8*half}
static __device__ __forceinline__ v16bf fragA(const unsigned short* tile, int lane) {
  int hl = (lane >> 4) & 1, m = lane & 15;
  const unsigned short* p = tile + m * LDA;
  v8bf lo = *(const v8bf*)(p + hl * 8);
  v8bf hi = *(const v8bf*)(p + 16 + hl * 8);
  return mk16(lo, hi);
}
// B fragment: 32x16 bf16, lane holds column n, K chunks {8*half, 16+8*half}
static __device__ __forceinline__ v16bf fragB(const unsigned short* tile, int lane) {
  int hl = (lane >> 4) & 1, n = lane & 15;
  const unsigned short* p = tile + n * LDA + hl * 8;
  v8bf lo = *(const v8bf*)(p);
  v8bf hi = *(const v8bf*)(p + 16);
  return mk16(lo, hi);
}
#define WMMA_BF16(a, b, c) \
  __builtin_amdgcn_wmma_f32_16x16x32_bf16(false, (a), false, (b), (short)0, (c), false, false)

// ---------- kernel 1: fp32 -> bf16 copy of x (shared-expert input) ----------
__global__ __launch_bounds__(256) void moe_cvt_kernel(const float* __restrict__ x,
                                                      unsigned int* __restrict__ xbf2,
                                                      long npairs) {
  long i = (long)blockIdx.x * 256 + threadIdx.x;
  if (i >= npairs) return;
  float2 f = ((const float2*)x)[i];
  xbf2[i] = pack2(f.x, f.y);
}

// ---------- kernel 2: gate + top-2 + dispatch ----------
__global__ __launch_bounds__(256) void moe_gate_kernel(
    const float* __restrict__ x, const float* __restrict__ gw,
    const float* __restrict__ gb, int* __restrict__ counts,
    int* __restrict__ tok, float* __restrict__ wt,
    unsigned short* __restrict__ xbuf, int T) {
  const int wv = threadIdx.x >> 5, lane = threadIdx.x & 31;
  const int t = blockIdx.x * 8 + wv;
  if (t >= T) return;
  const int e = lane & 15, hl = lane >> 4;

  const float* xr = x + (size_t)t * D_;
  const float* xh = xr + hl * 1024;
  const float* wr = gw + (size_t)e * D_ + hl * 1024;
  float acc = 0.f;
#pragma unroll 4
  for (int d = 0; d < 1024; d += 4) {
    float4 a = *(const float4*)(xh + d);
    float4 b = *(const float4*)(wr + d);
    acc += a.x * b.x + a.y * b.y + a.z * b.z + a.w * b.w;
  }
  acc += __shfl_xor(acc, 16, 32);                    // full-D logit on all lanes
  float sp = acc > 20.f ? acc : log1pf(__expf(acc)); // stable softplus
  float score = sqrtf(sp);
  float biased = score + gb[e];

  // top-1 (lowest index on tie, like top_k)
  float v = biased; int bi = e;
#pragma unroll
  for (int m = 1; m < 16; m <<= 1) {
    float ov = __shfl_xor(v, m, 32);
    int   oi = __shfl_xor(bi, m, 32);
    if (ov > v || (ov == v && oi < bi)) { v = ov; bi = oi; }
  }
  const int e1 = bi;
  float v2 = (e == e1) ? -1e30f : biased; int bi2 = e;
#pragma unroll
  for (int m = 1; m < 16; m <<= 1) {
    float ov = __shfl_xor(v2, m, 32);
    int   oi = __shfl_xor(bi2, m, 32);
    if (ov > v2 || (ov == v2 && oi < bi2)) { v2 = ov; bi2 = oi; }
  }
  const int e2 = bi2;
  const float s1 = __shfl(score, e1, 32);
  const float s2 = __shfl(score, e2, 32);
  const float inv = 1.f / (s1 + s2);
  const float w1 = s1 * inv, w2 = s2 * inv;

  int p1 = 0, p2 = 0;
  if (lane == 0) {
    p1 = atomicAdd(&counts[e1], 1);
    p2 = atomicAdd(&counts[e2], 1);
  }
  p1 = __shfl(p1, 0, 32);
  p2 = __shfl(p2, 0, 32);
  if (lane == 0) {
    if (p1 < CAP_) { tok[e1 * CAP_ + p1] = t; wt[e1 * CAP_ + p1] = w1; }
    if (p2 < CAP_) { tok[e2 * CAP_ + p2] = t; wt[e2 * CAP_ + p2] = w2; }
  }
  unsigned int* dst1 = (p1 < CAP_)
      ? (unsigned int*)xbuf + (((size_t)e1 * CAP_ + p1) * D_ >> 1) : nullptr;
  unsigned int* dst2 = (p2 < CAP_)
      ? (unsigned int*)xbuf + (((size_t)e2 * CAP_ + p2) * D_ >> 1) : nullptr;
  for (int i = lane; i < D_ / 2; i += 32) {
    float2 f = ((const float2*)xr)[i];
    unsigned int pk = pack2(f.x, f.y);
    if (dst1) dst1[i] = pk;
    if (dst2) dst2[i] = pk;
  }
}

// ---------- kernel 3: h = bf16( silu(X @ W1^T) * (X @ W3^T) ) ----------
// Block tile 128x64, 8 waves as 4x2 grid of 32x32 wave tiles (2x2 WMMA each).
__global__ __launch_bounds__(256) void moe_gateup_kernel(
    const unsigned short* __restrict__ Xall, const float* __restrict__ W1all,
    const float* __restrict__ W3all, unsigned short* __restrict__ Hall,
    const int* __restrict__ counts, int rowCap) {
  const int e = blockIdx.z;
  const int rowBase = blockIdx.y * 128;
  const int colBase = blockIdx.x * 64;
  if (counts) {
    int c = counts[e]; c = c > CAP_ ? CAP_ : c;
    if (rowBase >= c) return;                        // uniform early-exit
  }
  const unsigned short* X = Xall + (size_t)e * rowCap * D_;
  unsigned short* H = Hall + (size_t)e * rowCap * I_;
  const float* w1 = W1all + (size_t)e * I_ * D_;
  const float* w3 = W3all + (size_t)e * I_ * D_;

  __shared__ unsigned short As[128 * LDA];
  __shared__ unsigned short B1s[64 * LDA];
  __shared__ unsigned short B3s[64 * LDA];

  const int tid = threadIdx.x, lane = tid & 31, wv = tid >> 5;
  const int wm = wv & 3, wn = wv >> 2;

  const v8f zero = {0.f, 0.f, 0.f, 0.f, 0.f, 0.f, 0.f, 0.f};
  v8f accG[2][2] = {{zero, zero}, {zero, zero}};
  v8f accU[2][2] = {{zero, zero}, {zero, zero}};

  const int ar = tid >> 2;            // B-tile staging: 64 rows x 4 chunks
  const int ac = (tid & 3) << 3;

  for (int k0 = 0; k0 < D_; k0 += 32) {
    // stage A: 128x32 bf16, 512 16B-chunks over 256 threads.
    // Async path: straight global->LDS on ASYNCcnt, overlaps with the
    // VALU-heavy weight conversion below.
#pragma unroll
    for (int i = 0; i < 2; ++i) {
      int c = tid + i * 256;
      int r = c >> 2, cc = (c & 3) << 3;
      const unsigned short* src = X + (size_t)(rowBase + r) * D_ + k0 + cc;
#ifdef ASYNC_LDS
      async_copy16(src, &As[r * LDA + cc]);
#else
      *(uint4*)&As[r * LDA + cc] = *(const uint4*)src;
#endif
    }
    // stage B1/B3: fp32 weights -> bf16 in-register, one 16B chunk each
    {
      const size_t off = (size_t)(colBase + ar) * D_ + k0 + ac;
      *(uint4*)&B1s[ar * LDA + ac] = cvt8(w1 + off);
      *(uint4*)&B3s[ar * LDA + ac] = cvt8(w3 + off);
      if (k0 + 32 < D_) {
        __builtin_prefetch(w1 + off + 32, 0, 1);    // global_prefetch_b8
        __builtin_prefetch(w3 + off + 32, 0, 1);
      }
    }
#ifdef ASYNC_LDS
    async_wait0();
#endif
    __syncthreads();
    v16bf a0  = fragA(&As[(wm * 32 + 0) * LDA], lane);
    v16bf a1  = fragA(&As[(wm * 32 + 16) * LDA], lane);
    v16bf b10 = fragB(&B1s[(wn * 32 + 0) * LDA], lane);
    v16bf b11 = fragB(&B1s[(wn * 32 + 16) * LDA], lane);
    v16bf b30 = fragB(&B3s[(wn * 32 + 0) * LDA], lane);
    v16bf b31 = fragB(&B3s[(wn * 32 + 16) * LDA], lane);
    accG[0][0] = WMMA_BF16(a0, b10, accG[0][0]);
    accG[0][1] = WMMA_BF16(a0, b11, accG[0][1]);
    accG[1][0] = WMMA_BF16(a1, b10, accG[1][0]);
    accG[1][1] = WMMA_BF16(a1, b11, accG[1][1]);
    accU[0][0] = WMMA_BF16(a0, b30, accU[0][0]);
    accU[0][1] = WMMA_BF16(a0, b31, accU[0][1]);
    accU[1][0] = WMMA_BF16(a1, b30, accU[1][0]);
    accU[1][1] = WMMA_BF16(a1, b31, accU[1][1]);
    __syncthreads();
  }
  // fused SwiGLU epilogue; C/D layout: VGPR r -> M = r + 8*half, N = lane&15
  const int hl = lane >> 4, ln = lane & 15;
#pragma unroll
  for (int mi = 0; mi < 2; ++mi)
#pragma unroll
    for (int ni = 0; ni < 2; ++ni)
#pragma unroll
      for (int r = 0; r < 8; ++r) {
        int row = rowBase + wm * 32 + mi * 16 + hl * 8 + r;
        int col = colBase + wn * 32 + ni * 16 + ln;
        float g = accG[mi][ni][r];
        float u = accU[mi][ni][r];
        float h = (g / (1.f + __expf(-g))) * u;     // silu(g) * u
        H[(size_t)row * I_ + col] = (unsigned short)f2bf(h);
      }
}

// ---------- kernel 4: out = H @ W2^T ; store (shared) or weighted scatter (routed) ----
__global__ __launch_bounds__(256) void moe_down_kernel(
    const unsigned short* __restrict__ Hall, const float* __restrict__ W2all,
    float* __restrict__ Y, const int* __restrict__ counts,
    const int* __restrict__ tok, const float* __restrict__ wt, int rowCap) {
  const int e = blockIdx.z;
  const int rowBase = blockIdx.y * 128;
  const int colBase = blockIdx.x * 64;
  int cnt = rowCap;
  if (counts) {
    cnt = counts[e]; cnt = cnt > CAP_ ? CAP_ : cnt;
    if (rowBase >= cnt) return;
  }
  const unsigned short* H = Hall + (size_t)e * rowCap * I_;
  const float* w2 = W2all + (size_t)e * D_ * I_;

  __shared__ unsigned short As[128 * LDA];
  __shared__ unsigned short Bs[64 * LDA];

  const int tid = threadIdx.x, lane = tid & 31, wv = tid >> 5;
  const int wm = wv & 3, wn = wv >> 2;
  const v8f zero = {0.f, 0.f, 0.f, 0.f, 0.f, 0.f, 0.f, 0.f};
  v8f acc[2][2] = {{zero, zero}, {zero, zero}};
  const int ar = tid >> 2, ac = (tid & 3) << 3;

  for (int k0 = 0; k0 < I_; k0 += 32) {
#pragma unroll
    for (int i = 0; i < 2; ++i) {
      int c = tid + i * 256;
      int r = c >> 2, cc = (c & 3) << 3;
      const unsigned short* src = H + (size_t)(rowBase + r) * I_ + k0 + cc;
#ifdef ASYNC_LDS
      async_copy16(src, &As[r * LDA + cc]);
#else
      *(uint4*)&As[r * LDA + cc] = *(const uint4*)src;
#endif
    }
    {
      const size_t off = (size_t)(colBase + ar) * I_ + k0 + ac;
      *(uint4*)&Bs[ar * LDA + ac] = cvt8(w2 + off);
      if (k0 + 32 < I_) __builtin_prefetch(w2 + off + 32, 0, 1);
    }
#ifdef ASYNC_LDS
    async_wait0();
#endif
    __syncthreads();
    v16bf a0 = fragA(&As[(wm * 32 + 0) * LDA], lane);
    v16bf a1 = fragA(&As[(wm * 32 + 16) * LDA], lane);
    v16bf b0 = fragB(&Bs[(wn * 32 + 0) * LDA], lane);
    v16bf b1 = fragB(&Bs[(wn * 32 + 16) * LDA], lane);
    acc[0][0] = WMMA_BF16(a0, b0, acc[0][0]);
    acc[0][1] = WMMA_BF16(a0, b1, acc[0][1]);
    acc[1][0] = WMMA_BF16(a1, b0, acc[1][0]);
    acc[1][1] = WMMA_BF16(a1, b1, acc[1][1]);
    __syncthreads();
  }
  const int hl = lane >> 4, ln = lane & 15;
  if (!counts) {
    // shared expert: plain store, initializes all of Y
#pragma unroll
    for (int mi = 0; mi < 2; ++mi)
#pragma unroll
      for (int ni = 0; ni < 2; ++ni)
#pragma unroll
        for (int r = 0; r < 8; ++r) {
          int row = rowBase + wm * 32 + mi * 16 + hl * 8 + r;
          int col = colBase + wn * 32 + ni * 16 + ln;
          Y[(size_t)row * D_ + col] = acc[mi][ni][r];
        }
  } else {
    // routed expert: weighted atomic scatter back to tokens
#pragma unroll
    for (int mi = 0; mi < 2; ++mi)
#pragma unroll
      for (int ni = 0; ni < 2; ++ni)
#pragma unroll
        for (int r = 0; r < 8; ++r) {
          int slot = rowBase + wm * 32 + mi * 16 + hl * 8 + r;
          int col = colBase + wn * 32 + ni * 16 + ln;
          if (slot < cnt) {
            int   t = tok[e * CAP_ + slot];
            float w = wt[e * CAP_ + slot];
            __hip_atomic_fetch_add(&Y[(size_t)t * D_ + col], w * acc[mi][ni][r],
                                   __ATOMIC_RELAXED, __HIP_MEMORY_SCOPE_AGENT);
          }
        }
  }
}

// ---------- host launch ----------
extern "C" void kernel_launch(void* const* d_in, const int* in_sizes, int n_in,
                              void* d_out, int out_size, void* d_ws, size_t ws_size,
                              hipStream_t stream) {
  (void)n_in; (void)out_size; (void)ws_size;
  const float* x      = (const float*)d_in[0];
  const float* gate_w = (const float*)d_in[2];
  const float* gate_b = (const float*)d_in[3];
  const float* W1     = (const float*)d_in[4];
  const float* W3     = (const float*)d_in[5];
  const float* W2     = (const float*)d_in[6];
  const float* Ws1    = (const float*)d_in[7];
  const float* Ws3    = (const float*)d_in[8];
  const float* Ws2    = (const float*)d_in[9];
  float* Y = (float*)d_out;
  const int T = in_sizes[0] / D_;   // 4096

  char* p = (char*)d_ws;
  auto take = [&](size_t bytes) {
    char* r = p; p += (bytes + 255) & ~(size_t)255; return r;
  };
  int*            counts = (int*)take(E_ * sizeof(int));
  int*            tok    = (int*)take((size_t)E_ * CAP_ * sizeof(int));
  float*          wt     = (float*)take((size_t)E_ * CAP_ * sizeof(float));
  unsigned short* xbuf   = (unsigned short*)take((size_t)E_ * CAP_ * D_ * 2);
  unsigned short* xbf    = (unsigned short*)take((size_t)T * D_ * 2);
  unsigned short* hbuf   = (unsigned short*)take((size_t)E_ * CAP_ * I_ * 2);
  unsigned short* hs     = (unsigned short*)take((size_t)T * I_ * 2);

  (void)hipMemsetAsync(counts, 0, E_ * sizeof(int), stream);

  const long npairs = (long)T * (D_ / 2);
  moe_cvt_kernel<<<dim3((unsigned)((npairs + 255) / 256)), 256, 0, stream>>>(
      x, (unsigned int*)xbf, npairs);
  moe_gate_kernel<<<dim3((unsigned)((T + 7) / 8)), 256, 0, stream>>>(
      x, gate_w, gate_b, counts, tok, wt, xbuf, T);

  // shared expert (initializes Y via plain stores)
  moe_gateup_kernel<<<dim3(I_ / 64, T / 128, 1), 256, 0, stream>>>(
      xbf, Ws1, Ws3, hs, nullptr, T);
  moe_down_kernel<<<dim3(D_ / 64, T / 128, 1), 256, 0, stream>>>(
      hs, Ws2, Y, nullptr, nullptr, nullptr, T);

  // routed experts (atomic combine into Y)
  moe_gateup_kernel<<<dim3(I_ / 64, CAP_ / 128, E_), 256, 0, stream>>>(
      xbuf, W1, W3, hbuf, counts, CAP_);
  moe_down_kernel<<<dim3(D_ / 64, CAP_ / 128, E_), 256, 0, stream>>>(
      hbuf, W2, Y, counts, tok, wt, CAP_);
}
